// WFRPointCloudLoss_44444321579528
// MI455X (gfx1250) — matrix operand order repaired and checked by reading, not compiled, not run on hardware
//
#include <hip/hip_runtime.h>
#include <hip/hip_bf16.h>
#include <stdint.h>

// ---------------------------------------------------------------------------
// WFR point-cloud Sinkhorn loss for MI455X (gfx1250, wave32).
//
// Bandwidth-bound GEMV loop -> iterated kernel matrix kept in bf16 (67 MB)
// plus bf16 transpose (67 MB): both fit in the 192 MB L2. Per-iteration
// reductions via v_wmma_f32_16x16x32_bf16 with a column-replicated weight
// operand. Each wave batches 4 fragment loads (8 global_load_b128 +
// 8 ds_load_b128) before 4 WMMAs on 4 independent accumulators; scheduling
// enforced with sched_group_barrier so only one load-wait per 4 WMMAs and
// no WMMA->WMMA RAW chain. Final K is rebuilt in f32 from converged u,v.
// ---------------------------------------------------------------------------

#define B_      8
#define I_      2048
#define J_      2048
#define EPSF    0.01f
#define INV_EPS 100.0f
#define PCOEF   (1.0f / 1.01f)       // 1/(1+eps)
#define BIGF    1e20f
#define HUGEF   1e30f
#define SMALLF  1e-7f
#define DXF     (1.0f / 2048.0f)
#define DYF     (1.0f / 2048.0f)
#define SCALEF  4.0f                  // 4 * delta^2, delta = 1
#define NITER_  50
#define WAVES   8

// Scheduler grouping hints (no-op if builtin unavailable).
#if defined(__has_builtin)
#if __has_builtin(__builtin_amdgcn_sched_group_barrier)
#define SCHED_PIPELINE()                                   \
    do {                                                   \
        __builtin_amdgcn_sched_group_barrier(0x020, 8, 0); /* 8 VMEM reads */ \
        __builtin_amdgcn_sched_group_barrier(0x100, 8, 0); /* 8 DS reads   */ \
        __builtin_amdgcn_sched_group_barrier(0x008, 4, 0); /* 4 WMMA       */ \
    } while (0)
#endif
#endif
#ifndef SCHED_PIPELINE
#define SCHED_PIPELINE() do {} while (0)
#endif

typedef __attribute__((ext_vector_type(16))) __bf16 v16bf;
typedef __attribute__((ext_vector_type(8)))  float  v8f;

union FragU { unsigned int u[8]; v16bf v; };

__device__ __forceinline__ unsigned short f2bf(float f) {
    unsigned int x = __float_as_uint(f);
    unsigned int r = x + 0x7FFFu + ((x >> 16) & 1u);   // round-to-nearest-even
    return (unsigned short)(r >> 16);
}

__device__ __forceinline__ float clip_huge(float x) {
    return fminf(fmaxf(x, 0.0f), HUGEF);
}

// --- init small vectors: u = v = 0, b = 1 ----------------------------------
__global__ void k_zero_init(float* u, float* v, float* bvec, int n) {
    int t = blockIdx.x * blockDim.x + threadIdx.x;
    if (t < n) { u[t] = 0.0f; v[t] = 0.0f; bvec[t] = 1.0f; }
}

__global__ void k_zero_buf(float* p, int n) {
    int t = blockIdx.x * blockDim.x + threadIdx.x;
    if (t < n) p[t] = 0.0f;
}

__global__ void k_clear_max(unsigned int* maxab) {
    maxab[0] = 0u; maxab[1] = 0u;
}

// --- (re)build bf16 K and K^T from C, u, v ---------------------------------
// force != 0: always build. Otherwise only when an absorb triggered
// (max(a) > BIG or max(b) > BIG, read from workspace).
__global__ void k_buildK(const float* __restrict__ C,
                         const float* __restrict__ u,
                         const float* __restrict__ v,
                         unsigned short* __restrict__ Kb,
                         unsigned short* __restrict__ KbT,
                         int force,
                         const unsigned int* __restrict__ maxab) {
    if (!force) {
        float ma = __uint_as_float(maxab[0]);
        float mb = __uint_as_float(maxab[1]);
        if (!(ma > BIGF || mb > BIGF)) return;
    }
    int bb = blockIdx.z;
    int i0 = blockIdx.y << 5;
    int j0 = blockIdx.x << 5;
    int tx = threadIdx.x & 31;
    int ty = threadIdx.x >> 5;        // 8 row-lanes of threads

    __shared__ unsigned short tile[32][33];

    for (int r = ty; r < 32; r += 8) {
        int i = i0 + r, j = j0 + tx;
        size_t off = ((size_t)bb * I_ + i) * J_ + j;
        float c = C[off];
        float k = clip_huge(expf((u[(size_t)bb * I_ + i] +
                                  v[(size_t)bb * J_ + j] - c) * INV_EPS));
        unsigned short h = f2bf(k);
        Kb[off] = h;                  // coalesced row-major write
        tile[r][tx] = h;
    }
    __syncthreads();
    for (int r = ty; r < 32; r += 8) {
        // KbT[bb][j0+r][i0+tx]  -- coalesced via LDS transpose
        KbT[((size_t)bb * J_ + j0 + r) * I_ + i0 + tx] = tile[tx][r];
    }
}

// --- one Sinkhorn half-sweep: xout_i = clip((1/(exp(y_i)*s_i))^p) ----------
// s_i = sum_j Kmat[bb][i][j] * vin[bb][j] * scaleD, via bf16 WMMA.
// Used for both row pass (Kmat=Kb) and col pass (Kmat=KbT).
// Requires Jin % (32*WAVES*4) == 0 (holds: 2048 = 32*8*8).
__global__ void k_sweep(const unsigned short* __restrict__ Kmat, // [B][Iout][Jin] bf16
                        const float* __restrict__ vin,           // [B][Jin]
                        const float* __restrict__ yvec,          // [B][Iout]
                        float* __restrict__ xout,                // [B][Iout]
                        unsigned int* __restrict__ maxptr,
                        float scaleD, int Iout, int Jin) {
    int bb   = blockIdx.y;
    int i0   = blockIdx.x << 4;       // 16 rows per block
    int tid  = threadIdx.x;
    int wave = tid >> 5;
    int lane = tid & 31;
    int hi   = lane >> 4;
    int m16  = lane & 15;

    // stage weight vector (vin * scaleD) as bf16 in LDS
    __shared__ unsigned short wlds[J_];
    for (int idx = tid; idx < Jin; idx += 256)
        wlds[idx] = f2bf(vin[(size_t)bb * Jin + idx] * scaleD);
    __syncthreads();

    const unsigned short* rowptr =
        Kmat + ((size_t)bb * Iout + i0 + m16) * Jin;

    v8f acc[4] = {v8f{}, v8f{}, v8f{}, v8f{}};
    int nchunk = Jin >> 5;            // 32 K-elements per WMMA
    for (int base = wave; base < nchunk; base += WAVES * 4) {
        // Phase 1: batch-load 4 fragment pairs into distinct registers.
        FragU A[4], Bf[4];
#pragma unroll
        for (int q = 0; q < 4; ++q) {
            int j0 = (base + WAVES * q) << 5;
            // A operand: 16x32 bf16 tile; lane(m16,hi) holds row i0+m16,
            // k = {hi*8..hi*8+7, 16+hi*8..16+hi*8+7}
            const uint4 x0 = *reinterpret_cast<const uint4*>(rowptr + j0 + hi * 8);
            const uint4 x1 = *reinterpret_cast<const uint4*>(rowptr + j0 + 16 + hi * 8);
            A[q].u[0] = x0.x; A[q].u[1] = x0.y; A[q].u[2] = x0.z; A[q].u[3] = x0.w;
            A[q].u[4] = x1.x; A[q].u[5] = x1.y; A[q].u[6] = x1.z; A[q].u[7] = x1.w;
            // B operand: 32x16, B[k][n] = w[j0+k] replicated across columns;
            // lane holds k = {hi*8..hi*8+7, 16+hi*8..16+hi*8+7}.
            const unsigned int* wp =
                reinterpret_cast<const unsigned int*>(&wlds[j0 + hi * 8]);
#pragma unroll
            for (int e = 0; e < 8; ++e) Bf[q].u[e] = wp[e < 4 ? e : e + 4];
        }
        // Phase 2: 4 back-to-back WMMAs, independent D/A/B register sets.
#pragma unroll
        for (int q = 0; q < 4; ++q)
            acc[q] = __builtin_amdgcn_wmma_f32_16x16x32_bf16(
                false, A[q].v, false, Bf[q].v, (short)0, acc[q], false, false);

        // Enforce loads-then-WMMA grouping per iteration.
        SCHED_PIPELINE();
    }
    v8f accs = acc[0] + acc[1] + acc[2] + acc[3];

    // cross-wave reduction. All N columns of D are equal, so only the N==0
    // lanes (lane 0 -> rows m=0..7 in VGPRs 0..7, lane 16 -> rows 8..15)
    // carry distinct data. Divergence after the WMMAs is legal.
    __shared__ float red[WAVES][2][8];
    if (m16 == 0) {
#pragma unroll
        for (int e = 0; e < 8; ++e) red[wave][hi][e] = accs[e];
    }
    __syncthreads();

    if (tid < 16) {
        int m  = tid;
        int hs = (m < 8) ? 0 : 1;
        int vg = m & 7;
        float s = 0.0f;
#pragma unroll
        for (int w = 0; w < WAVES; ++w) s += red[w][hs][vg];
        float y = yvec[(size_t)bb * Iout + i0 + m];
        float t = 1.0f / (expf(y) * s);
        float x = clip_huge(powf(t, PCOEF));
        xout[(size_t)bb * Iout + i0 + m] = x;
        atomicMax(maxptr, __float_as_uint(x));   // x >= 0: bit-order == fp-order
    }
}

// --- apply absorb / carry b --------------------------------------------------
__global__ void k_apply(float* __restrict__ u, float* __restrict__ v,
                        const float* __restrict__ a,
                        const float* __restrict__ bnew,
                        float* __restrict__ bvec,
                        const unsigned int* __restrict__ maxab,
                        int force, int n) {
    int t = blockIdx.x * blockDim.x + threadIdx.x;
    if (t >= n) return;
    float ma = __uint_as_float(maxab[0]);
    float mb = __uint_as_float(maxab[1]);
    bool absorb = force || (ma > BIGF) || (mb > BIGF);
    if (absorb) {
        u[t] += EPSF * logf(a[t]);
        v[t] += EPSF * logf(bnew[t]);
        bvec[t] = 1.0f;
    } else {
        bvec[t] = bnew[t];
    }
}

// --- final f32 K + marginals + transport ------------------------------------
// 8 rows per block; column partials kept in registers, one atomicAdd per
// column per block.
__global__ void k_finalK(const float* __restrict__ C,
                         const float* __restrict__ u,
                         const float* __restrict__ v,
                         float* __restrict__ Kout,
                         float* __restrict__ margrow,
                         float* __restrict__ margcol,
                         float* __restrict__ transport) {
    int bb = blockIdx.y;
    int i0 = blockIdx.x << 3;
    int tid = threadIdx.x;
    __shared__ float sred[256];

    float colacc[8];
#pragma unroll
    for (int q = 0; q < 8; ++q) colacc[q] = 0.0f;
    float tsum = 0.0f;

    for (int r = 0; r < 8; ++r) {
        int i = i0 + r;
        float ui = u[(size_t)bb * I_ + i];
        size_t rowoff = ((size_t)bb * I_ + i) * J_;
        float rs = 0.0f;
#pragma unroll
        for (int q = 0; q < 8; ++q) {
            int j = tid + (q << 8);
            float c = C[rowoff + j];
            float k = clip_huge(expf((ui + v[(size_t)bb * J_ + j] - c) * INV_EPS));
            Kout[rowoff + j] = k;
            rs   += k;
            tsum += k * c;
            colacc[q] += k;
        }
        sred[tid] = rs; __syncthreads();
        for (int s = 128; s > 0; s >>= 1) {
            if (tid < s) sred[tid] += sred[tid + s];
            __syncthreads();
        }
        if (tid == 0) margrow[(size_t)bb * I_ + i] = sred[0] * DYF;
        __syncthreads();
    }
#pragma unroll
    for (int q = 0; q < 8; ++q) {
        int j = tid + (q << 8);
        atomicAdd(&margcol[(size_t)bb * J_ + j], colacc[q] * DXF);
    }
    sred[tid] = tsum; __syncthreads();
    for (int s = 128; s > 0; s >>= 1) {
        if (tid < s) sred[tid] += sred[tid + s];
        __syncthreads();
    }
    if (tid == 0) atomicAdd(&transport[bb], sred[0]);
}

// --- epilogue: KL penalties + outputs ---------------------------------------
__global__ void k_epilogue(const float* __restrict__ margrow,
                           const float* __restrict__ margcol,
                           const float* __restrict__ transport,
                           float* __restrict__ out) {
    int bb = blockIdx.x;
    int tid = threadIdx.x;
    __shared__ float sm[256];

    float acc1 = 0.0f;
    for (int i = tid; i < I_; i += 256) {
        float d = margrow[(size_t)bb * I_ + i] / (1.0f + SMALLF);
        acc1 += d * logf(d + SMALLF) - d + 1.0f;
    }
    float acc2 = 0.0f;
    for (int j = tid; j < J_; j += 256) {
        float d = margcol[(size_t)bb * J_ + j] / (1.0f + SMALLF);
        acc2 += d * logf(d + SMALLF) - d + 1.0f;
    }
    sm[tid] = acc1; __syncthreads();
    for (int s = 128; s > 0; s >>= 1) { if (tid < s) sm[tid] += sm[tid + s]; __syncthreads(); }
    float c1 = sm[0] * DXF;
    __syncthreads();
    sm[tid] = acc2; __syncthreads();
    for (int s = 128; s > 0; s >>= 1) { if (tid < s) sm[tid] += sm[tid + s]; __syncthreads(); }
    float c2 = sm[0] * DYF;
    if (tid == 0) {
        float tr  = SCALEF * transport[bb] * DXF * DYF;
        float con = SCALEF * (c1 + c2);
        out[bb]      = con + tr;   // p_opt
        out[8 + bb]  = tr;         // transport
        out[16 + bb] = con;        // constrain
    }
}

// ---------------------------------------------------------------------------
extern "C" void kernel_launch(void* const* d_in, const int* in_sizes, int n_in,
                              void* d_out, int out_size, void* d_ws, size_t ws_size,
                              hipStream_t stream) {
    (void)in_sizes; (void)n_in; (void)out_size; (void)ws_size;
    const float* C = (const float*)d_in[0];

    float* out  = (float*)d_out;
    float* Kout = out + 24;                       // [B,I,J] f32 output region

    // bf16 K + K^T scratch lives inside the (not-yet-needed) K output region:
    // 2 * B*I*J * 2 bytes == B*I*J * 4 bytes, 16-byte aligned at +96 B.
    unsigned short* Kb  = reinterpret_cast<unsigned short*>(Kout);
    unsigned short* KbT = Kb + (size_t)B_ * I_ * J_;

    const int BI = B_ * I_;                       // == B_*J_
    float* ws        = (float*)d_ws;              // ~460 KB used
    float* u         = ws;
    float* v         = u + BI;
    float* a         = v + BI;
    float* bvec      = a + BI;
    float* bnew      = bvec + BI;
    float* margrow   = bnew + BI;
    float* margcol   = margrow + BI;
    float* transport = margcol + BI;
    unsigned int* maxab = (unsigned int*)(transport + B_);   // [maxa, maxb]

    dim3 blk(256);
    dim3 gTile(J_ / 32, I_ / 32, B_);             // 64 x 64 x 8 for buildK
    dim3 gSweep(I_ / 16, B_);                     // 128 x 8
    dim3 gFinal(I_ / 8, B_);                      // 256 x 8

    k_zero_init<<<BI / 256, blk, 0, stream>>>(u, v, bvec, BI);
    k_buildK<<<gTile, blk, 0, stream>>>(C, u, v, Kb, KbT, 1, maxab);

    for (int it = 0; it < NITER_; ++it) {
        k_clear_max<<<1, 1, 0, stream>>>(maxab);
        // row pass: a from s = K b dy
        k_sweep<<<gSweep, blk, 0, stream>>>(Kb,  bvec, u, a,    maxab,     DYF, I_, J_);
        // col pass: b_new from s2 = K^T a dx
        k_sweep<<<gSweep, blk, 0, stream>>>(KbT, a,    v, bnew, maxab + 1, DXF, J_, I_);
        k_apply<<<BI / 256, blk, 0, stream>>>(u, v, a, bnew, bvec, maxab,
                                              (it == NITER_ - 1) ? 1 : 0, BI);
        if (it != NITER_ - 1)                     // last absorb folded into finalK
            k_buildK<<<gTile, blk, 0, stream>>>(C, u, v, Kb, KbT, 0, maxab);
    }

    k_zero_buf<<<(BI + B_ + 255) / 256, blk, 0, stream>>>(margcol, BI + B_);
    k_finalK<<<gFinal, blk, 0, stream>>>(C, u, v, Kout, margrow, margcol, transport);
    k_epilogue<<<B_, blk, 0, stream>>>(margrow, margcol, transport, out);
}
